// DCNConvModule_54589034332207
// MI455X (gfx1250) — matrix operand order repaired and checked by reading, not compiled, not run on hardware
//
#include <hip/hip_runtime.h>
#include <hip/hip_bf16.h>

typedef __attribute__((ext_vector_type(16))) __bf16 v16bf;
typedef __attribute__((ext_vector_type(8)))  float  v8f;

#define CIN   256
#define COUT  256
#define HH    128
#define WW    128
#define HWSZ  16384            // 128*128
#define KKTOT 2304             // CIN*9
#define PTILE 16               // output positions per workgroup
#define CBLK  32               // input channels per K block
#define KBLK  288              // CBLK*9
#define NGRP  32
#define CPG   8                // channels per group
#define GSIZE 131072           // CPG*HWSZ elements per (n,group)

// ---------------------------------------------------------------------------
// Kernel 1: fused deformable-im2col (bilinear gather -> LDS bf16) + WMMA GEMM
// ---------------------------------------------------------------------------
__global__ __launch_bounds__(256) void dcn_wmma_kernel(
    const float* __restrict__ x, const float* __restrict__ off,
    const __bf16* __restrict__ wt, const float* __restrict__ bias,
    float* __restrict__ y)
{
  __shared__ __bf16 s_cols[PTILE][KBLK + 8];   // padded rows vs bank conflicts
  __shared__ int    s_tidx[9][PTILE][4];
  __shared__ float  s_tw  [9][PTILE][4];

  const int tid  = threadIdx.x;
  const int lane = tid & 31;
  const int wv   = tid >> 5;                   // 8 waves

  const int g  = blockIdx.x;                   // 0..2047
  const int n  = g >> 10;                      // 1024 tiles per batch image
  const int p0 = (g & 1023) * PTILE;           // position base (16 | 128)
  const int h  = p0 / WW;
  const int w0 = p0 % WW;

  // ---- bilinear tap table: one thread per (tap, position) ----
  if (tid < 9 * PTILE) {
    const int t = tid / PTILE;
    const int p = tid % PTILE;
    const int sp = h * WW + (w0 + p);
    const float dy = off[(size_t)(n * 18 + 2 * t)     * HWSZ + sp];
    const float dx = off[(size_t)(n * 18 + 2 * t + 1) * HWSZ + sp];
    const float py = dy + (float)(t / 3) + (float)(h - 1);
    const float px = dx + (float)(t % 3) + (float)(w0 + p - 1);
    const float y0f = floorf(py), x0f = floorf(px);
    const int   y0  = (int)y0f,   x0  = (int)x0f;
    const float wy1 = py - y0f,   wx1 = px - x0f;
    const float wy0 = 1.f - wy1,  wx0 = 1.f - wx1;
    const float cw[4] = { wy0 * wx0, wy0 * wx1, wy1 * wx0, wy1 * wx1 };
    const int   cy[4] = { y0, y0, y0 + 1, y0 + 1 };
    const int   cx[4] = { x0, x0 + 1, x0, x0 + 1 };
#pragma unroll
    for (int c = 0; c < 4; ++c) {
      const bool ok = (cy[c] >= 0) && (cy[c] < HH) && (cx[c] >= 0) && (cx[c] < WW);
      const int yc = min(max(cy[c], 0), HH - 1);
      const int xc = min(max(cx[c], 0), WW - 1);
      s_tidx[t][p][c] = yc * WW + xc;
      s_tw  [t][p][c] = ok ? cw[c] : 0.f;
    }
  }
  __syncthreads();

  v8f acc0 = {};
  v8f acc1 = {};
  const int oc0 = wv * 16;              // M-tile 0 (oc base)
  const int oc1 = (wv + 8) * 16;        // M-tile 1

  const int p_s   = tid & 15;           // sampling: lane -> position (coalesced)
  const int pair0 = tid >> 4;           // 16 thread-groups over (c_local, tap)

  // A-layout (16-bit 16x32, wave32): lane holds K in {kb..kb+7} U {kb+16..kb+23}
  const int kb = (lane >> 4) * 8;
  // B-layout: same K order as A, lane holds K in {kb..kb+7} U {kb+16..kb+23}
  const int nb = (lane >> 4) * 16;
  const int mr = lane & 15;             // A row / B column within tile

  for (int cb = 0; cb < CIN / CBLK; ++cb) {
    // ---- deformable im2col: sample 32ch x 9taps x 16pos into LDS (bf16) ----
    for (int pi = pair0; pi < CBLK * 9; pi += 16) {
      const int cl = pi / 9;
      const int t  = pi % 9;
      const float* xc = x + (size_t)(n * CIN + cb * CBLK + cl) * HWSZ;
      const float v = s_tw[t][p_s][0] * xc[s_tidx[t][p_s][0]]
                    + s_tw[t][p_s][1] * xc[s_tidx[t][p_s][1]]
                    + s_tw[t][p_s][2] * xc[s_tidx[t][p_s][2]]
                    + s_tw[t][p_s][3] * xc[s_tidx[t][p_s][3]];
      s_cols[p_s][cl * 9 + t] = (__bf16)v;
    }
    __syncthreads();

    // ---- 9 WMMA K-steps over this 288-wide K block ----
#pragma unroll
    for (int s = 0; s < 9; ++s) {
      v16bf bfrag;
#pragma unroll
      for (int j = 0; j < 16; ++j)
        bfrag[j] = s_cols[mr][s * 32 + kb + (j < 8 ? j : j + 8)];

      const size_t kg = (size_t)cb * KBLK + s * 32 + kb;
      const __bf16* wr0 = wt + (size_t)(oc0 + mr) * KKTOT + kg;
      const __bf16* wr1 = wt + (size_t)(oc1 + mr) * KKTOT + kg;
      v16bf a0, a1;
#pragma unroll
      for (int j = 0; j < 8; ++j) {
        a0[j] = wr0[j];  a0[j + 8] = wr0[16 + j];
        a1[j] = wr1[j];  a1[j + 8] = wr1[16 + j];
      }
      acc0 = __builtin_amdgcn_wmma_f32_16x16x32_bf16(
                 false, a0, false, bfrag, (short)0, acc0, false, false);
      acc1 = __builtin_amdgcn_wmma_f32_16x16x32_bf16(
                 false, a1, false, bfrag, (short)0, acc1, false, false);
    }
    __syncthreads();
  }

  // ---- epilogue: bias + store conv output (f32) ----
  // C/D layout: element r of acc -> M = r + 8*(lane/16), N = lane%16
  const int col = lane & 15;
  const int mhi = (lane >> 4) * 8;
#pragma unroll
  for (int r = 0; r < 8; ++r) {
    const int ocA = oc0 + mhi + r;
    const int ocB = oc1 + mhi + r;
    y[(size_t)(n * COUT + ocA) * HWSZ + p0 + col] = acc0[r] + bias[ocA];
    y[(size_t)(n * COUT + ocB) * HWSZ + p0 + col] = acc1[r] + bias[ocB];
  }
}

// ---------------------------------------------------------------------------
// Helpers: weight f32->bf16, stats zeroing, GroupNorm reduction + apply
// ---------------------------------------------------------------------------
__global__ void wt_to_bf16(const float* __restrict__ w, __bf16* __restrict__ o,
                           int nElem) {
  const int i = blockIdx.x * blockDim.x + threadIdx.x;
  if (i < nElem) o[i] = (__bf16)w[i];
}

__global__ void zero_stats(float* s) { s[threadIdx.x] = 0.f; }

__global__ __launch_bounds__(256) void gn_stats(const float* __restrict__ y,
                                                float* __restrict__ st) {
  __shared__ float rs[256], rq[256];
  const int pair = blockIdx.x >> 3;    // (n,group) pair, 0..63
  const int part = blockIdx.x & 7;     // 8-way split
  const float4* base =
      (const float4*)(y + (size_t)pair * GSIZE + (size_t)part * (GSIZE / 8));
  float s = 0.f, q = 0.f;
  for (int i = threadIdx.x; i < (GSIZE / 8) / 4; i += 256) {
    const float4 v = base[i];
    s += v.x + v.y + v.z + v.w;
    q += v.x * v.x + v.y * v.y + v.z * v.z + v.w * v.w;
  }
  rs[threadIdx.x] = s; rq[threadIdx.x] = q;
  __syncthreads();
  for (int o = 128; o > 0; o >>= 1) {
    if (threadIdx.x < o) {
      rs[threadIdx.x] += rs[threadIdx.x + o];
      rq[threadIdx.x] += rq[threadIdx.x + o];
    }
    __syncthreads();
  }
  if (threadIdx.x == 0) {
    atomicAdd(&st[pair * 2],     rs[0]);
    atomicAdd(&st[pair * 2 + 1], rq[0]);
  }
}

__global__ __launch_bounds__(256) void gn_apply(
    const float* __restrict__ y, const float* __restrict__ st,
    const float* __restrict__ gamma, const float* __restrict__ beta,
    float* __restrict__ out) {
  const size_t e = ((size_t)blockIdx.x * 256 + threadIdx.x) * 4;
  const int ch   = (int)((e >> 14) & 255);
  const int n    = (int)(e >> 22);
  const int pair = n * NGRP + (ch >> 3);
  const float mu  = st[pair * 2]     * (1.f / (float)GSIZE);
  const float var = st[pair * 2 + 1] * (1.f / (float)GSIZE) - mu * mu;
  const float inv = rsqrtf(var + 1e-5f);
  const float ga  = gamma[ch] * inv;
  const float be  = beta[ch] - mu * ga;
  const float4 v = *(const float4*)(y + e);
  float4 o;
  o.x = fmaxf(v.x * ga + be, 0.f);
  o.y = fmaxf(v.y * ga + be, 0.f);
  o.z = fmaxf(v.z * ga + be, 0.f);
  o.w = fmaxf(v.w * ga + be, 0.f);
  *(float4*)(out + e) = o;
}

// ---------------------------------------------------------------------------
extern "C" void kernel_launch(void* const* d_in, const int* in_sizes, int n_in,
                              void* d_out, int out_size, void* d_ws, size_t ws_size,
                              hipStream_t stream) {
  (void)in_sizes; (void)n_in; (void)out_size; (void)ws_size;
  const float* x      = (const float*)d_in[0];
  const float* off    = (const float*)d_in[1];
  const float* weight = (const float*)d_in[2];
  const float* bias   = (const float*)d_in[3];
  const float* gamma  = (const float*)d_in[4];
  const float* beta   = (const float*)d_in[5];
  float* out = (float*)d_out;

  char* ws = (char*)d_ws;
  float*  y  = (float*)ws;                            // 33,554,432 B conv out
  __bf16* wt = (__bf16*)(ws + 33554432);              //  1,179,648 B bf16 W
  float*  st = (float*)(ws + 34734080);               //        512 B stats

  const int nw = COUT * KKTOT;                        // 589,824 weight elems
  wt_to_bf16<<<(nw + 255) / 256, 256, 0, stream>>>(weight, wt, nw);
  zero_stats<<<1, 128, 0, stream>>>(st);
  dcn_wmma_kernel<<<2048, 256, 0, stream>>>(x, off, wt, bias, y);
  gn_stats<<<512, 256, 0, stream>>>(y, st);
  gn_apply<<<8192, 256, 0, stream>>>(y, st, gamma, beta, out);
}
